// ComplexChannelAttn_7198365188195
// MI455X (gfx1250) — matrix operand order repaired and checked by reading, not compiled, not run on hardware
//
#include <hip/hip_runtime.h>
#include <hip/hip_bf16.h>

typedef __bf16 bf16;
typedef __attribute__((ext_vector_type(16))) __bf16 v16bf;
typedef __attribute__((ext_vector_type(8)))  __bf16 v8bf;
typedef __attribute__((ext_vector_type(8)))  float  v8f;

#define KSTEP 64   // K elements staged per pipeline stage (2 WMMA k-substeps)
#define LDSK2 72   // padded LDS K-stride (64 + 8): 144B rows, conflict-free

#define ACT_NONE  0
#define ACT_RELU  1
#define ACT_SIG   2
#define ACT_SCALE 3

// CDNA5 async global->LDS copy (ASYNCcnt-tracked, no staging VGPRs).
__device__ __forceinline__ void async_copy_b128(const bf16* gptr, bf16* lptr)
{
  unsigned           lds_off = (unsigned)(unsigned long long)(uintptr_t)lptr;
  unsigned long long gaddr   = (unsigned long long)(uintptr_t)gptr;
  asm volatile("global_load_async_to_lds_b128 %0, %1, off"
               :: "v"(lds_off), "v"(gaddr) : "memory");
}

__device__ __forceinline__ void wait_async0()
{
  asm volatile("s_wait_asynccnt 0x0" ::: "memory");
}

// ---------------------------------------------------------------------------
// Generic bf16 GEMM: C[M,N] = act( A[M,K] * Bt[N,K]^T ), f32 accumulation via
// v_wmma_f32_16x16x32_bf16. Block tile 128x128, 8 waves (2x4), wave tile 64x32.
// Double-buffered LDS; global->LDS staging via async copies overlapped with
// WMMA; one s_wait_asynccnt + barrier per K-stage.
// ---------------------------------------------------------------------------
template<int ACT, typename OutT>
__global__ __launch_bounds__(256, 2)
void gemm_bf16_kernel(const bf16* __restrict__ A, const bf16* __restrict__ Bt,
                      OutT* __restrict__ Cmat,
                      int Kdim, int lda, int ldb, int ldc,
                      long long strideA, long long strideB, long long strideC,
                      float scale)
{
  A    += (size_t)blockIdx.z * strideA;
  Bt   += (size_t)blockIdx.z * strideB;
  Cmat += (size_t)blockIdx.z * strideC;

  __shared__ bf16 ldsA[2][128 * LDSK2];
  __shared__ bf16 ldsB[2][128 * LDSK2];

  const int lane = threadIdx.x & 31;
  const int wave = threadIdx.x >> 5;
  const int wm   = wave >> 2;          // 0..1 : M wave tile
  const int wn   = wave & 3;           // 0..3 : N wave tile
  const int g    = lane >> 4;          // half-wave group
  const int r    = lane & 15;
  const int rowBlock = blockIdx.y * 128;
  const int colBlock = blockIdx.x * 128;

  v8f acc[4][2] = {};

  const int nK = Kdim / KSTEP;

  // Per-thread staging slots: 128 rows x 64 cols bf16 = 1024 b128 chunks per
  // matrix, 4 per thread. trow = idx>>3, tcol = (idx&7)*8 (K contiguous).
  // ---- prologue: async-stage tile 0 into buffer 0 ----
  #pragma unroll
  for (int it = 0; it < 4; ++it) {
    int idx  = (int)threadIdx.x + it * 256;
    int trow = idx >> 3;
    int tcol = (idx & 7) << 3;
    async_copy_b128(A  + (size_t)(rowBlock + trow) * lda + tcol,
                    &ldsA[0][trow * LDSK2 + tcol]);
    async_copy_b128(Bt + (size_t)(colBlock + trow) * ldb + tcol,
                    &ldsB[0][trow * LDSK2 + tcol]);
  }
  wait_async0();
  __syncthreads();

  for (int kt = 0; kt < nK; ++kt) {
    const int buf = kt & 1;

    // prefetch stage kt+2 into L2/WGP$ (global_prefetch_b8)
    if (kt + 2 < nK) {
      const int k2   = (kt + 2) * KSTEP;
      const int trow = (int)threadIdx.x >> 3;
      const int tcol = ((int)threadIdx.x & 7) << 3;
      __builtin_prefetch(A  + (size_t)(rowBlock + trow) * lda + k2 + tcol, 0, 1);
      __builtin_prefetch(Bt + (size_t)(colBlock + trow) * ldb + k2 + tcol, 0, 1);
    }

    // async-stage tile kt+1 into the other buffer (overlaps with WMMA below);
    // safe: every wave finished reading buf^1 before the barrier ending kt-1.
    if (kt + 1 < nK) {
      const int k1 = (kt + 1) * KSTEP;
      #pragma unroll
      for (int it = 0; it < 4; ++it) {
        int idx  = (int)threadIdx.x + it * 256;
        int trow = idx >> 3;
        int tcol = (idx & 7) << 3;
        async_copy_b128(A  + (size_t)(rowBlock + trow) * lda + k1 + tcol,
                        &ldsA[buf ^ 1][trow * LDSK2 + tcol]);
        async_copy_b128(Bt + (size_t)(colBlock + trow) * ldb + k1 + tcol,
                        &ldsB[buf ^ 1][trow * LDSK2 + tcol]);
      }
    }

    // ---- compute: 2 k-substeps x (4M x 2N) WMMAs from buffer `buf` ----
    #pragma unroll
    for (int ks = 0; ks < 2; ++ks) {
      v16bf afrag[4], bfrag[2];
      // B fragment: lane = column r, K = 16*g..16*g+15 contiguous
      #pragma unroll
      for (int jn = 0; jn < 2; ++jn) {
        const bf16* base = &ldsB[buf][(wn * 32 + jn * 16 + r) * LDSK2 + ks * 32];
        v8bf lo = *(const v8bf*)(base + g * 16);
        v8bf hi = *(const v8bf*)(base + g * 16 + 8);
        #pragma unroll
        for (int j = 0; j < 8; ++j) { bfrag[jn][j] = lo[j]; bfrag[jn][8 + j] = hi[j]; }
      }
      // A fragment: lane = row r, K = 16*g..16*g+15 contiguous
      #pragma unroll
      for (int i = 0; i < 4; ++i) {
        const bf16* base = &ldsA[buf][(wm * 64 + i * 16 + r) * LDSK2 + ks * 32];
        v8bf lo = *(const v8bf*)(base + g * 16);
        v8bf hi = *(const v8bf*)(base + g * 16 + 8);
        #pragma unroll
        for (int j = 0; j < 8; ++j) { afrag[i][j] = lo[j]; afrag[i][8 + j] = hi[j]; }
      }
      #pragma unroll
      for (int i = 0; i < 4; ++i)
        #pragma unroll
        for (int jn = 0; jn < 2; ++jn)
          acc[i][jn] = __builtin_amdgcn_wmma_f32_16x16x32_bf16(
              false, afrag[i], false, bfrag[jn], (short)0, acc[i][jn], false, false);
    }

    // ---- publish stage kt+1: wait own async copies, then barrier ----
    if (kt + 1 < nK) {
      wait_async0();
      __syncthreads();
    }
  }

  // Epilogue: C/D layout -> lane r is column, VGPR p is row p + 8*g.
  #pragma unroll
  for (int i = 0; i < 4; ++i) {
    #pragma unroll
    for (int jn = 0; jn < 2; ++jn) {
      const int col = colBlock + wn * 32 + jn * 16 + r;
      #pragma unroll
      for (int p = 0; p < 8; ++p) {
        const int row = rowBlock + wm * 64 + i * 16 + g * 8 + p;
        float v = acc[i][jn][p];
        if (ACT == ACT_RELU)  v = fmaxf(v, 0.0f);
        if (ACT == ACT_SIG)   v = 1.0f / (1.0f + __expf(-v));
        if (ACT == ACT_SCALE) v *= scale;
        Cmat[(size_t)row * ldc + col] = (OutT)v;
      }
    }
  }
}

// ---------------------------------------------------------------------------
// fp32 -> bf16 bulk convert
// ---------------------------------------------------------------------------
__global__ __launch_bounds__(256)
void f32_to_bf16_kernel(const float* __restrict__ in, bf16* __restrict__ out, int n)
{
  int i = blockIdx.x * blockDim.x + threadIdx.x;
  int stride = gridDim.x * blockDim.x;
  for (; i < n; i += stride) out[i] = (bf16)in[i];
}

// fp32 [rows][cols] -> bf16 [cols][rows] (weights are tiny; stays in L2)
__global__ __launch_bounds__(256)
void convert_transpose_kernel(const float* __restrict__ in, bf16* __restrict__ out,
                              int rows, int cols)
{
  int i = blockIdx.x * blockDim.x + threadIdx.x;
  int n = rows * cols;
  int stride = gridDim.x * blockDim.x;
  for (; i < n; i += stride) {
    int rr = i / cols, cc = i - rr * cols;
    out[(size_t)cc * rows + rr] = (bf16)in[i];
  }
}

// bf16 tiled transpose: in [rows][cols] -> out [cols][rows], batched in z
__global__ __launch_bounds__(256)
void transpose_bf16_kernel(const bf16* __restrict__ in, bf16* __restrict__ out,
                           int rows, int cols)
{
  __shared__ bf16 tile[32][33];
  const bf16* src = in  + (size_t)blockIdx.z * rows * cols;
  bf16*       dst = out + (size_t)blockIdx.z * rows * cols;
  int c0 = blockIdx.x * 32, r0 = blockIdx.y * 32;
  int tx = threadIdx.x, ty = threadIdx.y;   // 32 x 8
  #pragma unroll
  for (int i = 0; i < 32; i += 8)
    tile[ty + i][tx] = src[(size_t)(r0 + ty + i) * cols + c0 + tx];
  __syncthreads();
  #pragma unroll
  for (int i = 0; i < 32; i += 8)
    dst[(size_t)(c0 + ty + i) * rows + r0 + tx] = tile[tx][ty + i];
}

// ---------------------------------------------------------------------------
// Row softmax: S f32 [rows][512] -> P bf16. One wave per row (16 vals/lane).
// ---------------------------------------------------------------------------
__global__ __launch_bounds__(256)
void softmax_rows_kernel(const float* __restrict__ S, bf16* __restrict__ P, int cols)
{
  const int row  = blockIdx.x * 8 + (threadIdx.x >> 5);
  const int lane = threadIdx.x & 31;
  const float* s = S + (size_t)row * cols;

  float vals[16];
  float m = -INFINITY;
  #pragma unroll
  for (int c = 0; c < 4; ++c) {
    float4 d = *(const float4*)(s + c * 128 + lane * 4);
    vals[c * 4 + 0] = d.x; vals[c * 4 + 1] = d.y;
    vals[c * 4 + 2] = d.z; vals[c * 4 + 3] = d.w;
    m = fmaxf(m, fmaxf(fmaxf(d.x, d.y), fmaxf(d.z, d.w)));
  }
  #pragma unroll
  for (int off = 16; off > 0; off >>= 1) m = fmaxf(m, __shfl_xor(m, off, 32));

  float sum = 0.0f;
  #pragma unroll
  for (int j = 0; j < 16; ++j) { vals[j] = __expf(vals[j] - m); sum += vals[j]; }
  #pragma unroll
  for (int off = 16; off > 0; off >>= 1) sum += __shfl_xor(sum, off, 32);
  const float inv = 1.0f / sum;

  bf16* p = P + (size_t)row * cols;
  #pragma unroll
  for (int c = 0; c < 4; ++c)
    #pragma unroll
    for (int j = 0; j < 4; ++j)
      p[c * 128 + lane * 4 + j] = (bf16)(vals[c * 4 + j] * inv);
}

// ---------------------------------------------------------------------------
extern "C" void kernel_launch(void* const* d_in, const int* in_sizes, int n_in,
                              void* d_out, int out_size, void* d_ws, size_t ws_size,
                              hipStream_t stream)
{
  (void)in_sizes; (void)n_in; (void)out_size; (void)ws_size;

  const float* x   = (const float*)d_in[0];
  const float* w1s[3] = { (const float*)d_in[1], (const float*)d_in[3], (const float*)d_in[5] };
  const float* w2s[3] = { (const float*)d_in[2], (const float*)d_in[4], (const float*)d_in[6] };
  float* out = (float*)d_out;

  const int Nb = 32, C = 512, HW = 1024, R = 256;
  const size_t ROWS = (size_t)Nb * C;                 // 16384

  char* ws = (char*)d_ws;
  size_t off = 0;
  bf16* E = (bf16*)(ws + off); off += ROWS * HW * 2;  // 32MB bf16; later reused
  float* S = (float*)E;                               // as 32MB f32 scores
  bf16* w1T[3]; for (int i = 0; i < 3; ++i) { w1T[i] = (bf16*)(ws + off); off += (size_t)R * HW * 2; }
  bf16* w2T[3]; for (int i = 0; i < 3; ++i) { w2T[i] = (bf16*)(ws + off); off += (size_t)HW * R * 2; }
  bf16* Hb[3];  for (int i = 0; i < 3; ++i) { Hb[i]  = (bf16*)(ws + off); off += ROWS * R * 2; }
  bf16* QKV[3]; for (int i = 0; i < 3; ++i) { QKV[i] = (bf16*)(ws + off); off += ROWS * HW * 2; }
  bf16* P  = (bf16*)(ws + off); off += (size_t)Nb * C * C * 2;
  bf16* VT = (bf16*)(ws + off); off += (size_t)Nb * HW * C * 2;

  // 1. precision staging: e and weights to bf16 (weights pre-transposed to [N][K])
  f32_to_bf16_kernel<<<8192, 256, 0, stream>>>(x, E, (int)(ROWS * HW));
  for (int i = 0; i < 3; ++i) {
    convert_transpose_kernel<<<512, 256, 0, stream>>>(w1s[i], w1T[i], HW, R);
    convert_transpose_kernel<<<512, 256, 0, stream>>>(w2s[i], w2T[i], R, HW);
  }

  // 2. H = relu(E @ W1)       [16384 x 256], K=1024
  for (int i = 0; i < 3; ++i)
    gemm_bf16_kernel<ACT_RELU, bf16><<<dim3(R / 128, ROWS / 128, 1), 256, 0, stream>>>(
        E, w1T[i], Hb[i], HW, HW, HW, R, 0, 0, 0, 1.0f);

  // 3. Q/K/V = sigmoid(H @ W2) [16384 x 1024], K=256
  for (int i = 0; i < 3; ++i)
    gemm_bf16_kernel<ACT_SIG, bf16><<<dim3(HW / 128, ROWS / 128, 1), 256, 0, stream>>>(
        Hb[i], w2T[i], QKV[i], R, R, R, HW, 0, 0, 0, 1.0f);

  // 4. S = Q @ K^T / sqrt(C)   per batch [512 x 512], K=1024 (K rows are Bt rows)
  gemm_bf16_kernel<ACT_SCALE, float><<<dim3(C / 128, C / 128, Nb), 256, 0, stream>>>(
      QKV[0], QKV[1], S, HW, HW, HW, C,
      (long long)C * HW, (long long)C * HW, (long long)C * C,
      0.044194173824159216f /* 1/sqrt(512) */);

  // 5. P = softmax_rows(S) -> bf16
  softmax_rows_kernel<<<(Nb * C) / 8, 256, 0, stream>>>(S, P, C);

  // 6. V^T per batch so PV reuses the [N][K] B-tile path
  transpose_bf16_kernel<<<dim3(HW / 32, C / 32, Nb), dim3(32, 8), 0, stream>>>(
      QKV[2], VT, C, HW);

  // 7. O = P @ V               per batch [512 x 1024], K=512, f32 output
  gemm_bf16_kernel<ACT_NONE, float><<<dim3(HW / 128, C / 128, Nb), 256, 0, stream>>>(
      P, VT, out, C, C, C, HW,
      (long long)C * C, (long long)HW * C, (long long)C * HW, 1.0f);
}
